// DecoupleGCN_43095701848345
// MI455X (gfx1250) — matrix operand order repaired and checked by reading, not compiled, not run on hardware
//
#include <hip/hip_runtime.h>
#include <hip/hip_bf16.h>

typedef __attribute__((ext_vector_type(16))) __bf16 v16bf;
typedef __attribute__((ext_vector_type(8)))  __bf16 v8bf;
typedef __attribute__((ext_vector_type(4)))  __bf16 v4bf;
typedef __attribute__((ext_vector_type(8)))  float  v8f;
typedef __attribute__((ext_vector_type(4)))  unsigned int u32x4;
typedef __attribute__((ext_vector_type(8)))  int i32x8;
typedef __attribute__((ext_vector_type(4)))  int i32x4;

#define NNODES 100000
#define IN_DIM 256
#define HID    128
#define NCLS   40
#define NPAD   48   // 40 padded to 3 WMMA N-tiles
#define ECHUNK 32   // edges per 8-lane group in segmented spmm (avg dst-run = 16)

// LDS layout (dynamic shared, so byte offsets are known for the TDM descriptor)
#define SW2T_OFF 0                 // [48][128] bf16  = 12 KB  (TDM target)
#define SA_OFF   (12*1024)         // [64][256] bf16  = 32 KB  (feature tile)
#define SH_OFF   (44*1024)         // [64][128] bf16  = 16 KB  (relu(h1) tile)
#define SMEM_BYTES (60*1024)

__device__ __forceinline__ __bf16 f2bf(float f) {
  unsigned u = __builtin_bit_cast(unsigned, f);
  u += 0x7FFFu + ((u >> 16) & 1u);                 // round-to-nearest-even
  unsigned short s = (unsigned short)(u >> 16);
  return __builtin_bit_cast(__bf16, s);
}

// ---- one-off: convert W1 -> bf16 transposed [n][k], W2 -> bf16 transposed padded [48][128]
__global__ void prep_weights(const float* __restrict__ W1, const float* __restrict__ W2,
                             __bf16* __restrict__ W1t, __bf16* __restrict__ W2t) {
  int stride = gridDim.x * blockDim.x;
  for (int i = blockIdx.x * blockDim.x + threadIdx.x; i < HID * IN_DIM; i += stride) {
    int n = i / IN_DIM, k = i % IN_DIM;
    W1t[i] = f2bf(W1[k * HID + n]);
  }
  for (int i = blockIdx.x * blockDim.x + threadIdx.x; i < NPAD * HID; i += stride) {
    int n = i / HID, k = i % HID;
    W2t[i] = f2bf(n < NCLS ? W2[k * NCLS + n] : 0.0f);
  }
}

// ---- fused dense stage: h0 = relu(X@W1) @ W2, bf16 WMMA with f32 accumulate
// block = 128 threads = 4 waves; each wave computes 16 output rows.
__global__ __launch_bounds__(128)
void gemm_fused(const float* __restrict__ feat, const __bf16* __restrict__ W1t,
                const __bf16* __restrict__ W2t, float* __restrict__ h0, int Nn) {
  extern __shared__ char smem[];
  __bf16* sW2t = (__bf16*)(smem + SW2T_OFF);
  __bf16* sA   = (__bf16*)(smem + SA_OFF);
  __bf16* sH   = (__bf16*)(smem + SH_OFF);

  const int row0  = blockIdx.x * 64;
  const int wv    = threadIdx.x >> 5;
  const int lane  = threadIdx.x & 31;
  const int lrow  = lane & 15;
  const int khalf = lane >> 4;

  // ---- stage W2t [48][128] bf16 into LDS offset 0 via the Tensor Data Mover
#if __has_builtin(__builtin_amdgcn_tensor_load_to_lds)
  if (threadIdx.x < 32) {   // one wave issues the DMA (EXEC ignored by TDM)
    unsigned long long ga = (unsigned long long)(const void*)W2t;
    u32x4 g0;
    g0[0] = 1u;                                             // count=1, user descriptor
    g0[1] = (unsigned)SW2T_OFF;                             // lds_addr (bytes)
    g0[2] = (unsigned)(ga & 0xFFFFFFFFull);                 // global_addr[31:0]
    g0[3] = (unsigned)((ga >> 32) & 0x1FFFFFFull) | (2u << 30); // addr[56:32] | type=2
    i32x8 g1;
    g1[0] = (1 << 16);                 // workgroup_mask=0, data_size=1 (2 bytes)
    g1[1] = (int)(128u << 16);         // tensor_dim0[15:0]=128 in bits[31:16]
    g1[2] = (int)(48u << 16);          // tensor_dim0 hi=0 | tensor_dim1 lo=48
    g1[3] = (int)(128u << 16);         // tensor_dim1 hi=0 | tile_dim0=128
    g1[4] = 48;                        // tile_dim1=48, tile_dim2=0
    g1[5] = 128;                       // tensor_dim0_stride low 32 = 128 elems
    g1[6] = 0;                         // stride0 hi | stride1 lo
    g1[7] = 0;
    i32x4 z4; z4[0]=0; z4[1]=0; z4[2]=0; z4[3]=0;
    i32x8 z8; z8[0]=0; z8[1]=0; z8[2]=0; z8[3]=0; z8[4]=0; z8[5]=0; z8[6]=0; z8[7]=0;
    __builtin_amdgcn_tensor_load_to_lds(g0, g1, z4, z4, z8, 0);
  }
#else
  for (int i = threadIdx.x; i < NPAD * HID; i += 128) sW2t[i] = W2t[i];
#endif

  // ---- stage 64x256 feature tile, fp32 -> bf16, coalesced float4 loads
  for (int idx = threadIdx.x; idx < 64 * 64; idx += 128) {
    int r = idx >> 6;
    int c4 = idx & 63;
    float4 f; f.x = 0.f; f.y = 0.f; f.z = 0.f; f.w = 0.f;
    int gr = row0 + r;
    if (gr < Nn) f = *(const float4*)(feat + (size_t)gr * IN_DIM + c4 * 4);
    v4bf b; b[0] = f2bf(f.x); b[1] = f2bf(f.y); b[2] = f2bf(f.z); b[3] = f2bf(f.w);
    *(v4bf*)(sA + r * IN_DIM + c4 * 4) = b;
  }
#if __has_builtin(__builtin_amdgcn_tensor_load_to_lds)
  __builtin_amdgcn_s_wait_tensorcnt(0);
#endif
  __syncthreads();

  // ---- layer 1: acc1[16 rows][128 cols] = Atile @ W1  (8 N-tiles, K=256 in steps of 32)
  v8f acc1[8];
#pragma unroll
  for (int t = 0; t < 8; ++t)
#pragma unroll
    for (int r = 0; r < 8; ++r) acc1[t][r] = 0.0f;

  const __bf16* aBase = sA + (wv * 16 + lrow) * IN_DIM;
  for (int kk = 0; kk < IN_DIM; kk += 32) {
    union { v16bf v; v8bf h[2]; } af;
    af.v = *(const v16bf*)(aBase + kk + 16 * khalf);         // K = kk+16h .. +15
#pragma unroll
    for (int nt = 0; nt < 8; ++nt) {
      // B fragment: lane = col (n), 16 contiguous K values (W1t is [n][k])
      const __bf16* bp = W1t + (size_t)(nt * 16 + lrow) * IN_DIM + kk + 16 * khalf;
      v16bf bfrag = *(const v16bf*)bp;
      acc1[nt] = __builtin_amdgcn_wmma_f32_16x16x32_bf16(
          false, af.v, false, bfrag, (short)0, acc1[nt], false, false);
    }
  }

  // ---- relu + requantize to bf16 into sH (wave-private rows: no barrier needed)
#pragma unroll
  for (int nt = 0; nt < 8; ++nt)
#pragma unroll
    for (int r = 0; r < 8; ++r) {
      float vv = acc1[nt][r];
      vv = vv > 0.0f ? vv : 0.0f;
      int m = r + 8 * khalf;           // C layout: M = r + 8*(lane/16), N = lane%16
      sH[(wv * 16 + m) * HID + nt * 16 + lrow] = f2bf(vv);
    }

  // ---- layer 2: acc2[16 rows][48 cols] = relu(h1) @ W2  (3 N-tiles, K=128)
  v8f acc2[3];
#pragma unroll
  for (int t = 0; t < 3; ++t)
#pragma unroll
    for (int r = 0; r < 8; ++r) acc2[t][r] = 0.0f;

  const __bf16* hBase = sH + (wv * 16 + lrow) * HID;
  for (int kk = 0; kk < HID; kk += 32) {
    union { v16bf v; v8bf h[2]; } af;
    af.v = *(const v16bf*)(hBase + kk + 16 * khalf);
#pragma unroll
    for (int nt = 0; nt < 3; ++nt) {
      v16bf bfrag = *(const v16bf*)(sW2t + (nt * 16 + lrow) * HID + kk + 16 * khalf);
      acc2[nt] = __builtin_amdgcn_wmma_f32_16x16x32_bf16(
          false, af.v, false, bfrag, (short)0, acc2[nt], false, false);
    }
  }

  // ---- store h0 [N][40] f32 (drop pad cols 40..47)
#pragma unroll
  for (int nt = 0; nt < 3; ++nt)
#pragma unroll
    for (int r = 0; r < 8; ++r) {
      int m = r + 8 * khalf;
      int n = nt * 16 + lrow;
      int grow = row0 + wv * 16 + m;
      if (n < NCLS && grow < Nn) h0[(size_t)grow * NCLS + n] = acc2[nt][r];
    }
}

// ---- segmented SpMM: z[dst] += val * x[src], edge_dst sorted.
// Each 8-lane group walks ECHUNK consecutive edges; lane j owns columns [5j,5j+5).
// Register-accumulate while dst unchanged; atomics only at segment boundaries.
__global__ void spmm_seg(const int* __restrict__ src, const int* __restrict__ dst,
                         const float* __restrict__ val, const float* __restrict__ x,
                         float* __restrict__ z, int E) {
  long long T = (long long)blockIdx.x * blockDim.x + threadIdx.x;
  int g   = (int)(T >> 3);
  int sub = (int)(T & 7);
  int e0 = g * ECHUNK;
  if (e0 >= E) return;
  int eEnd = e0 + ECHUNK;
  if (eEnd > E) eEnd = E;

  float acc[5];
#pragma unroll
  for (int j = 0; j < 5; ++j) acc[j] = 0.0f;
  int cur = dst[e0];

  for (int e = e0; e < eEnd; ++e) {
    if (sub == 0 && e + 8 < E) {
      int s2 = __ldg(&src[e + 8]);
      __builtin_prefetch(x + (size_t)s2 * NCLS, 0, 0);   // global_prefetch_b8
    }
    int d = dst[e];                       // uniform across the 8-lane group
    if (d != cur) {                       // flush finished segment
      float* zr = z + (size_t)cur * NCLS + sub * 5;
#pragma unroll
      for (int j = 0; j < 5; ++j) {
        if (acc[j] != 0.0f) atomicAdd(&zr[j], acc[j]);
        acc[j] = 0.0f;
      }
      cur = d;
    }
    int s = src[e];
    float v = val[e];
    const float* xr = x + (size_t)s * NCLS + sub * 5;
#pragma unroll
    for (int j = 0; j < 5; ++j) acc[j] += v * xr[j];
  }
  // final flush
  {
    float* zr = z + (size_t)cur * NCLS + sub * 5;
#pragma unroll
    for (int j = 0; j < 5; ++j)
      if (acc[j] != 0.0f) atomicAdd(&zr[j], acc[j]);
  }
}

__global__ void zero_f32v4(float4* __restrict__ p, long long n4) {
  long long stride = (long long)gridDim.x * blockDim.x;
  float4 zq; zq.x = 0.f; zq.y = 0.f; zq.z = 0.f; zq.w = 0.f;
  for (long long i = (long long)blockIdx.x * blockDim.x + threadIdx.x; i < n4; i += stride)
    p[i] = zq;
}

extern "C" void kernel_launch(void* const* d_in, const int* in_sizes, int n_in,
                              void* d_out, int out_size, void* d_ws, size_t ws_size,
                              hipStream_t stream) {
  const float* feat = (const float*)d_in[0];
  const float* W1   = (const float*)d_in[1];
  const float* W2   = (const float*)d_in[2];
  const int*   esrc = (const int*)d_in[3];
  const int*   edst = (const int*)d_in[4];
  const float* eval = (const float*)d_in[5];
  const int E = in_sizes[3];
  const int N = NNODES;

  char* ws = (char*)d_ws;
  __bf16* W2t = (__bf16*)(ws);                 // 12 KB
  __bf16* W1t = (__bf16*)(ws + 12 * 1024);     // 64 KB
  float*  h0  = (float*)(ws + 76 * 1024);      // N*40 f32 = 16 MB
  float*  h1  = h0 + (size_t)N * NCLS;         // N*40 f32 = 16 MB
  float*  out = (float*)d_out;

  prep_weights<<<64, 256, 0, stream>>>(W1, W2, W1t, W2t);

  dim3 ggrid((N + 63) / 64);
  gemm_fused<<<ggrid, 128, SMEM_BYTES, stream>>>(feat, W1t, W2t, h0, N);

  long long hn  = (long long)N * NCLS;           // 4,000,000 (divisible by 4)
  long long hn4 = hn / 4;
  int ngroups = (E + ECHUNK - 1) / ECHUNK;
  int sblocks = (int)(((long long)ngroups * 8 + 255) / 256);

  zero_f32v4<<<2048, 256, 0, stream>>>((float4*)h1, hn4);
  spmm_seg<<<sblocks, 256, 0, stream>>>(esrc, edst, eval, h0, h1, E);

  zero_f32v4<<<2048, 256, 0, stream>>>((float4*)out, hn4);
  spmm_seg<<<sblocks, 256, 0, stream>>>(esrc, edst, eval, h1, out, E);
}
